// MyViT_13993003451074
// MI455X (gfx1250) — matrix operand order repaired and checked by reading, not compiled, not run on hardware
//
#include <hip/hip_runtime.h>
#include <hip/hip_bf16.h>

// ---------------- model constants ----------------
constexpr int BATCH   = 256;
constexpr int SEQ     = 65;     // 64 patches + cls
constexpr int SP      = 80;     // SEQ padded to 5x16 for WMMA M/N tiles
constexpr int SP2     = 96;     // SEQ padded to 3x32 for WMMA K tiles
constexpr int D       = 768;
constexpr int NH      = 12;
constexpr int DH      = 64;
constexpr int MLPD    = 3072;
constexpr int LAY     = 12;
constexpr int OUTD    = 1000;
constexpr int OUTP    = 1024;   // head N padded to 16x64
constexpr int IND     = 16;     // C*PATCH
constexpr int R       = BATCH * SEQ;   // 16640 rows (divisible by 128)

typedef __attribute__((ext_vector_type(16))) _Float16 v16h;
typedef __attribute__((ext_vector_type(8)))  float    v8f;

// ---------------- WMMA helpers (wave32, gfx1250) ----------------
__device__ inline v8f wmma_f16(v16h a, v16h b, v8f c) {
  return __builtin_amdgcn_wmma_f32_16x16x32_f16(
      /*neg_a=*/false, a, /*neg_b=*/false, b,
      /*c_mod=*/(short)0, c, /*reuse_a=*/false, /*reuse_b=*/false);
}

// A fragment 16x32 (MxK), row-major source, ld in halves.
__device__ inline v16h frag_a(const _Float16* p, int ld) {
  int lane = threadIdx.x & 31;
  int m  = lane & 15;
  int ko = (lane < 16) ? 0 : 16;
  v16h r;
#pragma unroll
  for (int i = 0; i < 8; ++i) {
    r[i]     = p[m * ld + ko + i];
    r[8 + i] = p[m * ld + ko + 8 + i];
  }
  return r;
}
// B fragment 32x16 (KxN), source stored [k][n].
__device__ inline v16h frag_b(const _Float16* p, int ld) {
  int lane = threadIdx.x & 31;
  int n  = lane & 15;
  int kb = (lane < 16) ? 0 : 16;
  v16h r;
#pragma unroll
  for (int i = 0; i < 16; ++i) r[i] = p[(kb + i) * ld + n];
  return r;
}
// B fragment, source stored transposed [n][k].
__device__ inline v16h frag_b_t(const _Float16* p, int ld) {
  int lane = threadIdx.x & 31;
  int n  = lane & 15;
  int kb = (lane < 16) ? 0 : 16;
  v16h r;
#pragma unroll
  for (int i = 0; i < 16; ++i) r[i] = p[n * ld + kb + i];
  return r;
}

// ---------------- CDNA5 async global->LDS copy (ASYNCcnt path) ----------------
// generic pointer to __shared__ has the LDS byte offset in its low 32 bits
__device__ inline unsigned lds_lo32(const void* p) {
  return (unsigned)(unsigned long long)p;
}
__device__ inline void async_b128(unsigned lds_off, const _Float16* g) {
  asm volatile("global_load_async_to_lds_b128 %0, %1, off"
               :: "v"(lds_off), "v"(g) : "memory");
}
__device__ inline void wait_async0() {
  asm volatile("s_wait_asynccnt 0x0" ::: "memory");
}

// ---------------- block reductions (256 threads) ----------------
__device__ inline float block_sum(float v, float* red) {
  int tid = threadIdx.x;
  red[tid] = v; __syncthreads();
  for (int off = 128; off > 0; off >>= 1) {
    if (tid < off) red[tid] += red[tid + off];
    __syncthreads();
  }
  float r = red[0]; __syncthreads();
  return r;
}
__device__ inline float block_max(float v, float* red) {
  int tid = threadIdx.x;
  red[tid] = v; __syncthreads();
  for (int off = 128; off > 0; off >>= 1) {
    if (tid < off) red[tid] = fmaxf(red[tid], red[tid + off]);
    __syncthreads();
  }
  float r = red[0]; __syncthreads();
  return r;
}

// ---------------- 0) fp32 -> fp16 weight conversion (one-shot) ----------------
__global__ __launch_bounds__(256)
void cvt_kernel(const float* __restrict__ src, _Float16* __restrict__ dst,
                long nvalid, long ntotal) {
  long stride = (long)gridDim.x * 256;
  for (long i = (long)blockIdx.x * 256 + threadIdx.x; i < ntotal; i += stride)
    dst[i] = (i < nvalid) ? (_Float16)src[i] : (_Float16)0;
}

// ---------------- 1) patch embed + cls + pos ----------------
__global__ __launch_bounds__(256)
void patch_embed_kernel(const float* __restrict__ images,
                        const float* __restrict__ lw, const float* __restrict__ lb,
                        const float* __restrict__ cls, const float* __restrict__ pos,
                        float* __restrict__ x) {
  int s = blockIdx.x, n = blockIdx.y, tid = threadIdx.x;
  __shared__ float sp[IND];
  if (s > 0 && tid < IND) sp[tid] = images[(long)n * 1024 + (long)(s - 1) * IND + tid];
  __syncthreads();
  for (int d = tid; d < D; d += 256) {
    float acc;
    if (s == 0) {
      acc = cls[d];
    } else {
      acc = lb[d];
      const float* wr = lw + (long)d * IND;
#pragma unroll
      for (int i = 0; i < IND; ++i) acc += sp[i] * wr[i];
    }
    x[((long)n * SEQ + s) * D + d] = acc + pos[(long)s * D + d];
  }
}

// ---------------- 2) layernorm (f32 in -> f16 out) ----------------
__global__ __launch_bounds__(256)
void ln_kernel(const float* __restrict__ x, const float* __restrict__ g,
               const float* __restrict__ b, _Float16* __restrict__ out) {
  __shared__ float red[256];
  long row = blockIdx.x;
  int tid = threadIdx.x;
  const float* xr = x + row * D;
  float v0 = xr[tid], v1 = xr[tid + 256], v2 = xr[tid + 512];
  float s  = block_sum(v0 + v1 + v2, red);
  float m  = s * (1.0f / D);
  float d0 = v0 - m, d1 = v1 - m, d2 = v2 - m;
  float ss = block_sum(d0 * d0 + d1 * d1 + d2 * d2, red);
  float inv = rsqrtf(ss * (1.0f / D) + 1e-5f);
  _Float16* o = out + row * D;
  o[tid]       = (_Float16)(d0 * inv * g[tid]       + b[tid]);
  o[tid + 256] = (_Float16)(d1 * inv * g[tid + 256] + b[tid + 256]);
  o[tid + 512] = (_Float16)(d2 * inv * g[tid + 512] + b[tid + 512]);
}

// ---------------- 3) per-head QKV projection, async-staged ----------------
// out[r, head*64+e] = sum_d h[r, head*64+d] * w16[head, e, d] + bias[head, e]
__global__ __launch_bounds__(128)
void qkv_kernel(const _Float16* __restrict__ h, const _Float16* __restrict__ w16,
                const float* __restrict__ bias, _Float16* __restrict__ out) {
  __shared__ _Float16 At[64 * 64];   // [m][d]
  __shared__ _Float16 Bt[64 * 64];   // [e][d]
  int tid = threadIdx.x;
  int wave = tid >> 5, lane = tid & 31;
  int head = blockIdx.y;
  long m0 = (long)blockIdx.x * 64;

  { // A: thread t -> row t>>1, 32 halves at (t&1)*32  (4 x b128)
    int mm = tid >> 1, hf = (tid & 1) * 32;
    const _Float16* ga = h + (m0 + mm) * D + head * DH + hf;
    unsigned la = lds_lo32(&At[mm * 64 + hf]);
#pragma unroll
    for (int i = 0; i < 4; ++i) async_b128(la + i * 16, ga + i * 8);
  }
  { // B: thread t -> 32 contiguous halves (4 x b128)
    const _Float16* gb = w16 + (long)head * DH * DH + tid * 32;
    unsigned lb = lds_lo32(&Bt[tid * 32]);
#pragma unroll
    for (int i = 0; i < 4; ++i) async_b128(lb + i * 16, gb + i * 8);
  }
  wait_async0();
  __syncthreads();

  v8f acc[4] = {};
#pragma unroll
  for (int ks = 0; ks < 2; ++ks) {
    v16h a = frag_a(&At[(wave * 16) * 64 + ks * 32], 64);
#pragma unroll
    for (int nt = 0; nt < 4; ++nt) {
      v16h bfr = frag_b_t(&Bt[(nt * 16) * 64 + ks * 32], 64);
      acc[nt] = wmma_f16(a, bfr, acc[nt]);
    }
  }

  int nl = lane & 15, mo = (lane < 16) ? 0 : 8;
#pragma unroll
  for (int nt = 0; nt < 4; ++nt) {
    int e = nt * 16 + nl;
    float bs = bias[head * DH + e];
#pragma unroll
    for (int r8 = 0; r8 < 8; ++r8) {
      long row = m0 + wave * 16 + r8 + mo;
      out[row * D + head * DH + e] = (_Float16)(acc[nt][r8] + bs);
    }
  }
}

// ---------------- 4) fused attention per (batch, head); residual into x ----------------
__global__ __launch_bounds__(160)
void attn_kernel(const _Float16* __restrict__ q, const _Float16* __restrict__ k,
                 const _Float16* __restrict__ v, float* __restrict__ x) {
  __shared__ __align__(16) char smem[SP * SP * 4 + 2 * SP * DH * 2 + SP2 * DH * 2];
  float*    Ssc = (float*)smem;                                      // 25600 B
  _Float16* Qs  = (_Float16*)(smem + SP * SP * 4);                   // 10240 B
  _Float16* Ks  = Qs + SP * DH;                                      // 10240 B
  _Float16* Af  = (_Float16*)(smem + SP * SP * 4);                   // aliases Qs/Ks
  _Float16* Vs  = (_Float16*)(smem + SP * SP * 4 + 2 * SP * DH * 2); // 12288 B

  int tid = threadIdx.x;                 // 160 = 5 waves
  int wave = tid >> 5, lane = tid & 31;
  int head = blockIdx.x, b = blockIdx.y;
  long base = (long)b * SEQ;

  for (int i = tid; i < SP * DH; i += 160) {
    int s = i >> 6, e = i & 63;
    _Float16 qv = (_Float16)0, kv = (_Float16)0;
    if (s < SEQ) {
      qv = q[(base + s) * D + head * DH + e];
      kv = k[(base + s) * D + head * DH + e];
    }
    Qs[i] = qv; Ks[i] = kv;
  }
  for (int i = tid; i < SP2 * DH; i += 160) {
    int s = i >> 6, e = i & 63;
    Vs[i] = (s < SEQ) ? v[(base + s) * D + head * DH + e] : (_Float16)0;
  }
  __syncthreads();

  { // scores = Q K^T / 8
    v8f acc[5] = {};
#pragma unroll
    for (int ks = 0; ks < 2; ++ks) {
      v16h a = frag_a(&Qs[(wave * 16) * DH + ks * 32], DH);
#pragma unroll
      for (int nt = 0; nt < 5; ++nt) {
        v16h bfr = frag_b_t(&Ks[(nt * 16) * DH + ks * 32], DH);
        acc[nt] = wmma_f16(a, bfr, acc[nt]);
      }
    }
    int nl = lane & 15, mo = (lane < 16) ? 0 : 8;
#pragma unroll
    for (int nt = 0; nt < 5; ++nt)
#pragma unroll
      for (int r8 = 0; r8 < 8; ++r8) {
        int m = wave * 16 + r8 + mo, t = nt * 16 + nl;
        Ssc[m * SP + t] = acc[nt][r8] * 0.125f;
      }
  }
  __syncthreads();

  if (tid < SP) { // softmax rows; Af aliases dead Qs/Ks
    int s = tid;
    if (s < SEQ) {
      float mx = -3.0e38f;
      for (int t = 0; t < SEQ; ++t) mx = fmaxf(mx, Ssc[s * SP + t]);
      float sum = 0.f;
      for (int t = 0; t < SEQ; ++t) sum += expf(Ssc[s * SP + t] - mx);
      float inv = 1.0f / sum;
      for (int t = 0; t < SP2; ++t)
        Af[s * SP2 + t] = (t < SEQ) ? (_Float16)(expf(Ssc[s * SP + t] - mx) * inv)
                                    : (_Float16)0;
    } else {
      for (int t = 0; t < SP2; ++t) Af[s * SP2 + t] = (_Float16)0;
    }
  }
  __syncthreads();

  { // O = Af(80x96) * V(96x64)
    v8f acc[4] = {};
#pragma unroll
    for (int ks = 0; ks < 3; ++ks) {
      v16h a = frag_a(&Af[(wave * 16) * SP2 + ks * 32], SP2);
#pragma unroll
      for (int nt = 0; nt < 4; ++nt) {
        v16h bfr = frag_b(&Vs[(ks * 32) * DH + nt * 16], DH);
        acc[nt] = wmma_f16(a, bfr, acc[nt]);
      }
    }
    int nl = lane & 15, mo = (lane < 16) ? 0 : 8;
#pragma unroll
    for (int nt = 0; nt < 4; ++nt)
#pragma unroll
      for (int r8 = 0; r8 < 8; ++r8) {
        int m = wave * 16 + r8 + mo;
        if (m < SEQ) {
          int e = nt * 16 + nl;
          x[(base + m) * D + head * DH + e] += acc[nt][r8];
        }
      }
  }
}

// ---------------- 5) WMMA GEMM, async double-buffered ----------------
// out = A(f16)[M,K] x B(f16)[N,K]^T.  Block: 128 thr, tile M=128, N=64.
// Each wave: 32x64 strip -> 2 A-frags x 4 B-frags = 8 WMMA / K-step.
// EPI 0: +bias, exact GELU -> f16   (FC1)
// EPI 1: +bias, residual add f32    (FC2)
// EPI 2: +bias, store f32           (head; Nvalid masks ragged N)
template <int EPI>
__global__ __launch_bounds__(128)
void gemm_wmma(const _Float16* __restrict__ A, const _Float16* __restrict__ B,
               const float* __restrict__ bias, _Float16* __restrict__ out16,
               float* __restrict__ out32, int K, int ldOut, int Nvalid) {
  __shared__ _Float16 At[2][128 * 32];   // [m][k] double-buffered
  __shared__ _Float16 Bt[2][64 * 32];    // [n][k] double-buffered
  int tid = threadIdx.x;
  int wave = tid >> 5, lane = tid & 31;
  long m0 = (long)blockIdx.x * 128;
  long n0 = (long)blockIdx.y * 64;

  // async staging: A -> thread t copies row t (32 halves, 4xb128);
  //                B -> thread t copies row t>>1 half (t&1) (16 halves, 2xb128)
  auto stage = [&](int buf, int ks) {
    const _Float16* ga = A + (m0 + tid) * K + ks * 32;
    unsigned la = lds_lo32(&At[buf][tid * 32]);
#pragma unroll
    for (int i = 0; i < 4; ++i) async_b128(la + i * 16, ga + i * 8);
    int n = tid >> 1, hf = (tid & 1) * 16;
    const _Float16* gb = B + (n0 + n) * K + ks * 32 + hf;
    unsigned lb = lds_lo32(&Bt[buf][n * 32 + hf]);
    async_b128(lb, gb);
    async_b128(lb + 16, gb + 8);
  };

  v8f acc0[4] = {}, acc1[4] = {};
  int nsteps = K >> 5;
  int cur = 0;
  stage(0, 0);
  for (int ks = 0; ks < nsteps; ++ks) {
    wait_async0();
    __syncthreads();
    if (ks + 1 < nsteps) stage(cur ^ 1, ks + 1);   // overlap next copy with math
    const _Float16* Ab = &At[cur][0];
    const _Float16* Bb = &Bt[cur][0];
    v16h a0 = frag_a(Ab + (wave * 32) * 32, 32);
    v16h a1 = frag_a(Ab + (wave * 32 + 16) * 32, 32);
#pragma unroll
    for (int nt = 0; nt < 4; ++nt) {
      v16h bfr = frag_b_t(Bb + (nt * 16) * 32, 32);
      acc0[nt] = wmma_f16(a0, bfr, acc0[nt]);
      acc1[nt] = wmma_f16(a1, bfr, acc1[nt]);
    }
    cur ^= 1;
  }

  int nl = lane & 15, mo = (lane < 16) ? 0 : 8;
#pragma unroll
  for (int nt = 0; nt < 4; ++nt) {
    long col = n0 + nt * 16 + nl;
    if (col >= Nvalid) continue;
    float bs = bias[col];
#pragma unroll
    for (int sub = 0; sub < 2; ++sub) {
      const v8f& acc = sub ? acc1[nt] : acc0[nt];
#pragma unroll
      for (int r8 = 0; r8 < 8; ++r8) {
        long row = m0 + wave * 32 + sub * 16 + r8 + mo;
        float val = acc[r8] + bs;
        if (EPI == 0) {
          float g = 0.5f * val * (1.0f + erff(val * 0.70710678f));
          out16[row * (long)ldOut + col] = (_Float16)g;
        } else if (EPI == 1) {
          out32[row * (long)ldOut + col] += val;
        } else {
          out32[row * (long)ldOut + col] = val;
        }
      }
    }
  }
}

// ---------------- 6) cls row extraction (f32 -> f16) ----------------
__global__ __launch_bounds__(256)
void cls_extract_kernel(const float* __restrict__ x, _Float16* __restrict__ cls16) {
  long n = blockIdx.x;
  int tid = threadIdx.x;
  for (int d = tid; d < D; d += 256)
    cls16[n * D + d] = (_Float16)x[(n * SEQ) * D + d];
}

// ---------------- 7) head softmax ----------------
__global__ __launch_bounds__(256)
void softmax_kernel(const float* __restrict__ logits, float* __restrict__ out) {
  __shared__ float red[256];
  long row = blockIdx.x;
  int tid = threadIdx.x;
  const float* lr = logits + row * OUTD;
  float mx = -3.0e38f;
  for (int c = tid; c < OUTD; c += 256) mx = fmaxf(mx, lr[c]);
  mx = block_max(mx, red);
  float s = 0.f;
  for (int c = tid; c < OUTD; c += 256) s += expf(lr[c] - mx);
  s = block_sum(s, red);
  float inv = 1.0f / s;
  for (int c = tid; c < OUTD; c += 256)
    out[row * OUTD + c] = expf(lr[c] - mx) * inv;
}

// ---------------- host launch ----------------
extern "C" void kernel_launch(void* const* d_in, const int* in_sizes, int n_in,
                              void* d_out, int out_size, void* d_ws, size_t ws_size,
                              hipStream_t stream) {
  const float* images   = (const float*)d_in[0];
  const float* linear_w = (const float*)d_in[1];
  const float* linear_b = (const float*)d_in[2];
  const float* cls_tok  = (const float*)d_in[3];
  const float* pos_emb  = (const float*)d_in[4];
  const float* ln1_g    = (const float*)d_in[5];
  const float* ln1_b    = (const float*)d_in[6];
  const float* wq       = (const float*)d_in[7];
  const float* bq       = (const float*)d_in[8];
  const float* wk       = (const float*)d_in[9];
  const float* bk       = (const float*)d_in[10];
  const float* wv       = (const float*)d_in[11];
  const float* bv       = (const float*)d_in[12];
  const float* ln2_g    = (const float*)d_in[13];
  const float* ln2_b    = (const float*)d_in[14];
  const float* w1       = (const float*)d_in[15];
  const float* b1       = (const float*)d_in[16];
  const float* w2       = (const float*)d_in[17];
  const float* b2       = (const float*)d_in[18];
  const float* head_w   = (const float*)d_in[19];
  const float* head_b   = (const float*)d_in[20];
  (void)in_sizes; (void)n_in; (void)out_size; (void)ws_size;

  constexpr long NWQ  = (long)LAY * NH * DH * DH;     // 589,824
  constexpr long NW1  = (long)LAY * MLPD * D;         // 28,311,552
  constexpr long NHW  = (long)OUTD * D;               // 768,000
  constexpr long NHWP = (long)OUTP * D;               // 786,432 (padded)

  // workspace carve-up
  char* p = (char*)d_ws;
  float*    x    = (float*)p;     p += (size_t)R * D * sizeof(float);
  _Float16* h    = (_Float16*)p;  p += (size_t)R * D * sizeof(_Float16);
  _Float16* q16  = (_Float16*)p;  p += (size_t)R * D * sizeof(_Float16);
  _Float16* k16  = (_Float16*)p;  p += (size_t)R * D * sizeof(_Float16);
  _Float16* v16  = (_Float16*)p;  p += (size_t)R * D * sizeof(_Float16);
  _Float16* m16  = (_Float16*)p;  p += (size_t)R * MLPD * sizeof(_Float16);
  _Float16* c16  = (_Float16*)p;  p += (size_t)BATCH * D * sizeof(_Float16);
  float*    lg   = (float*)p;     p += (size_t)BATCH * OUTD * sizeof(float);
  _Float16* wq16 = (_Float16*)p;  p += (size_t)NWQ * sizeof(_Float16);
  _Float16* wk16 = (_Float16*)p;  p += (size_t)NWQ * sizeof(_Float16);
  _Float16* wv16 = (_Float16*)p;  p += (size_t)NWQ * sizeof(_Float16);
  _Float16* w116 = (_Float16*)p;  p += (size_t)NW1 * sizeof(_Float16);
  _Float16* w216 = (_Float16*)p;  p += (size_t)NW1 * sizeof(_Float16);
  _Float16* hw16 = (_Float16*)p;  p += (size_t)NHWP * sizeof(_Float16);

  // 0) one-shot fp32 -> fp16 weight conversion
  cvt_kernel<<<1024, 256, 0, stream>>>(wq, wq16, NWQ, NWQ);
  cvt_kernel<<<1024, 256, 0, stream>>>(wk, wk16, NWQ, NWQ);
  cvt_kernel<<<1024, 256, 0, stream>>>(wv, wv16, NWQ, NWQ);
  cvt_kernel<<<4096, 256, 0, stream>>>(w1, w116, NW1, NW1);
  cvt_kernel<<<4096, 256, 0, stream>>>(w2, w216, NW1, NW1);
  cvt_kernel<<<1024, 256, 0, stream>>>(head_w, hw16, NHW, NHWP);

  // 1) patch embed + cls + pos
  patch_embed_kernel<<<dim3(SEQ, BATCH), 256, 0, stream>>>(
      images, linear_w, linear_b, cls_tok, pos_emb, x);

  // 2) transformer layers
  for (int l = 0; l < LAY; ++l) {
    ln_kernel<<<R, 256, 0, stream>>>(x, ln1_g + (long)l * D, ln1_b + (long)l * D, h);

    long woff = (long)l * NH * DH * DH, boff = (long)l * NH * DH;
    qkv_kernel<<<dim3(R / 64, NH), 128, 0, stream>>>(h, wq16 + woff, bq + boff, q16);
    qkv_kernel<<<dim3(R / 64, NH), 128, 0, stream>>>(h, wk16 + woff, bk + boff, k16);
    qkv_kernel<<<dim3(R / 64, NH), 128, 0, stream>>>(h, wv16 + woff, bv + boff, v16);

    attn_kernel<<<dim3(NH, BATCH), 160, 0, stream>>>(q16, k16, v16, x);

    ln_kernel<<<R, 256, 0, stream>>>(x, ln2_g + (long)l * D, ln2_b + (long)l * D, h);

    gemm_wmma<0><<<dim3(R / 128, MLPD / 64), 128, 0, stream>>>(
        h, w116 + (long)l * MLPD * D, b1 + (long)l * MLPD, m16, nullptr,
        D, MLPD, MLPD);
    gemm_wmma<1><<<dim3(R / 128, D / 64), 128, 0, stream>>>(
        m16, w216 + (long)l * D * MLPD, b2 + (long)l * D, nullptr, x,
        MLPD, D, D);
  }

  // 3) classification head + softmax
  cls_extract_kernel<<<BATCH, 256, 0, stream>>>(x, c16);
  gemm_wmma<2><<<dim3(BATCH / 128, OUTP / 64), 128, 0, stream>>>(
      c16, hw16, head_b, nullptr, lg, D, OUTD, OUTD);
  softmax_kernel<<<BATCH, 256, 0, stream>>>(lg, (float*)d_out);
}
